// MConv_17575006175640
// MI455X (gfx1250) — matrix operand order repaired and checked by reading, not compiled, not run on hardware
//
#include <hip/hip_runtime.h>

typedef _Float16 half_t;
typedef _Float16 v16h __attribute__((ext_vector_type(16)));
typedef float    v8f  __attribute__((ext_vector_type(8)));

// ---------------- workspace layout (bytes) ----------------
#define XB_OFF    0ULL                         // 4 * 8*64*64*128 f32 = 64 MB (NHWC per branch)
#define SAMP_OFF  67108864ULL                  // 32768 * 1152 f16   = 75.5 MB (per-branch, reused)
#define SBUF_OFF  142606336ULL                 // 32768 * 512  f16   = 32 MB  (post BN/SiLU)
#define OFFS_OFF  176160768ULL                 // 8*18*64*64 f32     = 2.36 MB (per-branch, reused)
#define DWF16_OFF 178520064ULL                 // 4 * 128*1152 f16
#define WP_OFF    179699712ULL                 // 4 * 9*128*20 f32 (offset weights, padded o->20)
#define FWF16_OFF 180068352ULL                 // 256*512 f16

__device__ inline v8f zero8() {
  v8f z;
#pragma unroll
  for (int i = 0; i < 8; ++i) z[i] = 0.0f;
  return z;
}

// ---------------- K0: NCHW -> 4 branch NHWC tensors ----------------
__global__ void split_nhwc(const float* __restrict__ x, float* __restrict__ xb) {
  long i = (long)blockIdx.x * 256 + threadIdx.x;   // 16,777,216 total
  int c  = (int)(i & 127);
  int w  = (int)((i >> 7) & 63);
  int h  = (int)((i >> 13) & 63);
  int b  = (int)((i >> 19) & 7);
  int br = (int)(i >> 22);
  int hh = 2*h + (br & 1);
  int ww = 2*w + (br >> 1);
  xb[i] = x[(((long)(b*128 + c) * 128) + hh) * 128 + ww];
}

// ---------------- weight prep ----------------
// deform weights: dw[o][c][ky][kx] -> f16 A[o][kk*128 + c]
__global__ void prep_dw(const float* __restrict__ dw, half_t* __restrict__ A) {
  int i = blockIdx.x * 256 + threadIdx.x;          // 147456
  if (i >= 128 * 1152) return;
  int o  = i / 1152;
  int kp = i - o * 1152;
  int kk = kp >> 7;
  int c  = kp & 127;
  A[i] = (half_t)dw[(o * 128 + c) * 9 + kk];
}
// offset weights: ow[o][c][ky][kx] -> wp[kk][c][o'] with o' padded to 20
__global__ void prep_offw(const float* __restrict__ ow, float* __restrict__ wp) {
  int i = blockIdx.x * 256 + threadIdx.x;          // 23040
  if (i >= 9 * 128 * 20) return;
  int o  = i % 20;
  int c  = (i / 20) % 128;
  int kk = i / (20 * 128);
  wp[i] = (o < 18) ? ow[(o * 128 + c) * 9 + kk] : 0.0f;
}
__global__ void prep_fw(const float* __restrict__ fw, half_t* __restrict__ F) {
  int i = blockIdx.x * 256 + threadIdx.x;          // 131072
  if (i < 256 * 512) F[i] = (half_t)fw[i];
}

// ---------------- K1: offset conv (fp32, LDS-staged weights) ----------------
__global__ void offconv(const float* __restrict__ xb, const float* __restrict__ wp,
                        const float* __restrict__ ob, float* __restrict__ offs) {
  __shared__ float wl[128 * 20];                   // 10 KB, staged per kk
  int tid = threadIdx.x;
  int pix = blockIdx.x * 256 + tid;                // 0..32767
  int b = pix >> 12, hw = pix & 4095, h = hw >> 6, w = hw & 63;
  float4 acc[5];
#pragma unroll
  for (int q = 0; q < 5; ++q) acc[q] = make_float4(0.f, 0.f, 0.f, 0.f);

  for (int kk = 0; kk < 9; ++kk) {
    __syncthreads();
    for (int j = tid; j < 128 * 20; j += 256) wl[j] = wp[kk * 128 * 20 + j];
    __syncthreads();
    int y  = h - 1 + kk / 3;
    int xx = w - 1 + kk % 3;
    if (y < 0 || y >= 64 || xx < 0 || xx >= 64) continue;
    const float* src = xb + (((long)(b * 64 + y) * 64 + xx) * 128);
    for (int c = 0; c < 128; c += 4) {
      float4 xv = *(const float4*)(src + c);
      const float* xs = (const float*)&xv;
#pragma unroll
      for (int d = 0; d < 4; ++d) {
        float s = xs[d];
        const float4* w4 = (const float4*)&wl[(c + d) * 20];
#pragma unroll
        for (int q = 0; q < 5; ++q) {
          float4 wv = w4[q];
          acc[q].x += s * wv.x; acc[q].y += s * wv.y;
          acc[q].z += s * wv.z; acc[q].w += s * wv.w;
        }
      }
    }
  }
  const float* accf = (const float*)acc;
#pragma unroll
  for (int o = 0; o < 18; ++o)
    offs[(long)(b * 18 + o) * 4096 + hw] = accf[o] + ob[o];
}

// ---------------- K2: bilinear sampler ----------------
typedef _Float16 v4h __attribute__((ext_vector_type(4)));

__device__ inline void add_corner(float4& acc, const float* __restrict__ base,
                                  int yi, int xi, float wgt) {
  if (yi >= 0 && yi < 64 && xi >= 0 && xi < 64) {
    float4 v = *(const float4*)(base + ((long)(yi * 64 + xi)) * 128);
    acc.x += wgt * v.x; acc.y += wgt * v.y; acc.z += wgt * v.z; acc.w += wgt * v.w;
  }
}

__global__ void sample_k(const float* __restrict__ xb, const float* __restrict__ offs,
                         half_t* __restrict__ samp) {
  int wave = (blockIdx.x * 256 + threadIdx.x) >> 5;  // 294912 tasks exactly
  int lane = threadIdx.x & 31;
  int pix = wave / 9;
  int kk  = wave - pix * 9;
  int b = pix >> 12, hw = pix & 4095, h = hw >> 6, w = hw & 63;
  float off0 = offs[(long)(b * 18 + kk * 2 + 0) * 4096 + hw];
  float off1 = offs[(long)(b * 18 + kk * 2 + 1) * 4096 + hw];
  float py = (float)(h - 1 + kk / 3) + off0;
  float px = (float)(w - 1 + kk % 3) + off1;
  float y0f = floorf(py), x0f = floorf(px);
  int   y0  = (int)y0f,  x0  = (int)x0f;
  float wy = py - y0f, wx = px - x0f;
  float w00 = (1.f - wy) * (1.f - wx), w01 = (1.f - wy) * wx;
  float w10 = wy * (1.f - wx),         w11 = wy * wx;
  int c = lane * 4;
  const float* base = xb + (long)b * (64 * 64 * 128) + c;
  float4 acc = make_float4(0.f, 0.f, 0.f, 0.f);
  add_corner(acc, base, y0,     x0,     w00);
  add_corner(acc, base, y0,     x0 + 1, w01);
  add_corner(acc, base, y0 + 1, x0,     w10);
  add_corner(acc, base, y0 + 1, x0 + 1, w11);
  v4h r; r[0] = (half_t)acc.x; r[1] = (half_t)acc.y; r[2] = (half_t)acc.z; r[3] = (half_t)acc.w;
  *(v4h*)(samp + (long)pix * 1152 + kk * 128 + c) = r;
}

// ---------------- K3/K4: WMMA GEMM, tile 128(M) x 64(N), K chunks of 32 ----------------
// A: f16 [M][Kdim] row-major (weights). B: f16 [Ntot][Kdim] row-major (acts).
// mode 0: epilogue BN-residual + SiLU -> f16 outH[n*512 + chanBase + m]
// mode 1: plain -> f32 outF NCHW (256 channels)
__global__ void gemm_wmma(const half_t* __restrict__ A, const half_t* __restrict__ B,
                          int Kdim, int mode, int chanBase,
                          half_t* __restrict__ outH, float* __restrict__ outF,
                          const float* __restrict__ gamma, const float* __restrict__ beta,
                          const float* __restrict__ mean, const float* __restrict__ var) {
  __shared__ half_t As[128 * 32];
  __shared__ half_t Bs[64 * 32];
  int tid  = threadIdx.x;
  int lane = tid & 31, wave = tid >> 5;
  int n0   = blockIdx.x * 64;
  int m0g  = blockIdx.y * 128;

  v8f acc[4];
#pragma unroll
  for (int j = 0; j < 4; ++j) acc[j] = zero8();

  int m   = wave * 16 + (lane & 15);
  int hiA = (lane >> 4) << 3;   // lanes 16-31 take K +16
  int hiB = (lane >> 4) << 3;   // lanes 16-31 take K +16

  for (int k0 = 0; k0 < Kdim; k0 += 32) {
    __syncthreads();
    {   // stage A chunk: 128 rows x 32 halfs
      int row = tid >> 1;
      int kof = (tid & 1) * 16;
      const uint4* src = (const uint4*)(A + (long)(m0g + row) * Kdim + k0 + kof);
      uint4* dst = (uint4*)(As + row * 32 + kof);
      dst[0] = src[0];
      dst[1] = src[1];
    }
    {   // stage B chunk: 64 rows x 32 halfs
      int row = tid >> 2;
      int kof = (tid & 3) * 8;
      *(uint4*)(Bs + row * 32 + kof) =
          *(const uint4*)(B + (long)(n0 + row) * Kdim + k0 + kof);
    }
    __syncthreads();

    union { unsigned u[8]; v16h v; } af;
    const unsigned* Ar = (const unsigned*)(As + m * 32);
#pragma unroll
    for (int i = 0; i < 8; ++i) af.u[i] = Ar[hiA + i];

#pragma unroll
    for (int j = 0; j < 4; ++j) {
      union { unsigned u[8]; v16h v; } bf;
      const unsigned* Br = (const unsigned*)(Bs + (j * 16 + (lane & 15)) * 32);
#pragma unroll
      for (int i = 0; i < 8; ++i) bf.u[i] = Br[hiB + i];
      acc[j] = __builtin_amdgcn_wmma_f32_16x16x32_f16(
          false, af.v, false, bf.v, (short)0, acc[j], false, false);
    }
  }

  int mrBase = wave * 16 + ((lane >> 4) << 3);
  int nl = lane & 15;
  if (mode == 0) {
#pragma unroll
    for (int r = 0; r < 8; ++r) {
      int mm = m0g + mrBase + r;
      int ch = chanBase + mm;
      float sc = gamma[ch] * rsqrtf(var[ch] + 1e-5f);
      float mu = mean[ch], bt = beta[ch];
#pragma unroll
      for (int j = 0; j < 4; ++j) {
        int n = n0 + j * 16 + nl;
        float y  = acc[j][r];
        float y2 = y + (y - mu) * sc + bt;
        float s  = y2 / (1.0f + expf(-y2));
        outH[(long)n * 512 + ch] = (half_t)s;
      }
    }
  } else {
#pragma unroll
    for (int r = 0; r < 8; ++r) {
      int mm = m0g + mrBase + r;
#pragma unroll
      for (int j = 0; j < 4; ++j) {
        int n = n0 + j * 16 + nl;
        outF[((long)(n >> 12) * 256 + mm) * 4096 + (n & 4095)] = acc[j][r];
      }
    }
  }
}

// ---------------- launch ----------------
extern "C" void kernel_launch(void* const* d_in, const int* in_sizes, int n_in,
                              void* d_out, int out_size, void* d_ws, size_t ws_size,
                              hipStream_t stream) {
  const float* x        = (const float*)d_in[0];
  const float* bn_gamma = (const float*)d_in[13];
  const float* bn_beta  = (const float*)d_in[14];
  const float* bn_mean  = (const float*)d_in[15];
  const float* bn_var   = (const float*)d_in[16];
  const float* final_w  = (const float*)d_in[17];

  char* ws = (char*)d_ws;
  float*  XB    = (float*) (ws + XB_OFF);
  half_t* SAMP  = (half_t*)(ws + SAMP_OFF);
  half_t* SBUF  = (half_t*)(ws + SBUF_OFF);
  float*  OFFS  = (float*) (ws + OFFS_OFF);
  half_t* DWF16 = (half_t*)(ws + DWF16_OFF);
  float*  WP    = (float*) (ws + WP_OFF);
  half_t* FWF16 = (half_t*)(ws + FWF16_OFF);

  split_nhwc<<<65536, 256, 0, stream>>>(x, XB);
  prep_fw<<<512, 256, 0, stream>>>(final_w, FWF16);
  for (int br = 0; br < 4; ++br) {
    prep_dw  <<<576, 256, 0, stream>>>((const float*)d_in[3 + 3 * br], DWF16 + (size_t)br * 128 * 1152);
    prep_offw<<<90,  256, 0, stream>>>((const float*)d_in[1 + 3 * br], WP + (size_t)br * 9 * 128 * 20);
  }

  for (int br = 0; br < 4; ++br) {
    const float* xbB = XB + (size_t)br * 8 * 64 * 64 * 128;
    offconv <<<128,   256, 0, stream>>>(xbB, WP + (size_t)br * 9 * 128 * 20,
                                        (const float*)d_in[2 + 3 * br], OFFS);
    sample_k<<<36864, 256, 0, stream>>>(xbB, OFFS, SAMP);
    gemm_wmma<<<dim3(512, 1), 256, 0, stream>>>(
        DWF16 + (size_t)br * 128 * 1152, SAMP, 1152,
        /*mode=*/0, /*chanBase=*/br * 128, SBUF, nullptr,
        bn_gamma, bn_beta, bn_mean, bn_var);
  }

  gemm_wmma<<<dim3(512, 2), 256, 0, stream>>>(
      FWF16, SBUF, 512, /*mode=*/1, 0, nullptr, (float*)d_out,
      nullptr, nullptr, nullptr, nullptr);
}
